// EncoderBlock_46840913330858
// MI455X (gfx1250) — matrix operand order repaired and checked by reading, not compiled, not run on hardware
//
#include <hip/hip_runtime.h>
#include <stdint.h>

typedef __bf16 bf16_t;
typedef __attribute__((ext_vector_type(16))) __bf16 v16bf;
typedef __attribute__((ext_vector_type(8)))  __bf16 v8bf;
typedef __attribute__((ext_vector_type(8)))  float  v8f;

// Problem constants (match reference)
#define BB   4
#define SS   1024
#define DD   1024
#define HH   16
#define DKK  64
#define DFFF 4096
#define MROWS (BB * SS)   // 4096

// ---------------------------------------------------------------------------
// helpers
// ---------------------------------------------------------------------------
__device__ inline v16bf cat8(v8bf lo, v8bf hi) {
  return __builtin_shufflevector(lo, hi, 0, 1, 2, 3, 4, 5, 6, 7,
                                 8, 9, 10, 11, 12, 13, 14, 15);
}

// Async DMA: global (16B) -> LDS, tracked by ASYNCcnt (CDNA5 §15.18.3 op 98)
__device__ inline void async_copy_b128(uint32_t lds_off, const void* gaddr) {
  asm volatile("global_load_async_to_lds_b128 %0, %1, off"
               :: "v"(lds_off), "v"(gaddr) : "memory");
}
__device__ inline void wait_asynccnt0() {
  asm volatile("s_wait_asynccnt 0x0" ::: "memory");
}

// ---------------------------------------------------------------------------
// fp32 -> bf16 conversion (weights / any tensor)
// ---------------------------------------------------------------------------
__global__ __launch_bounds__(256) void cvt_f32_bf16_kernel(
    const float* __restrict__ in, bf16_t* __restrict__ out, int n) {
  int i = blockIdx.x * blockDim.x + threadIdx.x;
  if (i < n) out[i] = (bf16_t)in[i];
}

// ---------------------------------------------------------------------------
// LayerNorm (torch-style: ddof=1, eps added to std, scalar alpha/beta)
// ---------------------------------------------------------------------------
__global__ __launch_bounds__(256) void layernorm_kernel(
    const float* __restrict__ x, const float* __restrict__ alpha,
    const float* __restrict__ beta, bf16_t* __restrict__ out) {
  const int row = blockIdx.x;
  const float* xr = x + (size_t)row * DD;
  float s = 0.f, sq = 0.f;
  for (int i = threadIdx.x; i < DD; i += 256) {
    float v = xr[i];
    s += v; sq += v * v;
  }
#pragma unroll
  for (int off = 16; off > 0; off >>= 1) {
    s  += __shfl_down(s,  off, 32);
    sq += __shfl_down(sq, off, 32);
  }
  __shared__ float ps[8], pq[8];
  const int wid = threadIdx.x >> 5;
  if ((threadIdx.x & 31) == 0) { ps[wid] = s; pq[wid] = sq; }
  __syncthreads();
  if (threadIdx.x == 0) {
    float ts = 0.f, tq = 0.f;
#pragma unroll
    for (int i = 0; i < 8; ++i) { ts += ps[i]; tq += pq[i]; }
    ps[0] = ts; pq[0] = tq;
  }
  __syncthreads();
  const float mean = ps[0] * (1.0f / DD);
  const float var  = (pq[0] - (float)DD * mean * mean) * (1.0f / (DD - 1));
  const float inv  = 1.0f / (sqrtf(fmaxf(var, 0.f)) + 1e-6f);
  const float a = alpha[0], b = beta[0];
  for (int i = threadIdx.x; i < DD; i += 256)
    out[(size_t)row * DD + i] = (bf16_t)(a * (xr[i] - mean) * inv + b);
}

// ---------------------------------------------------------------------------
// bf16 WMMA GEMM: out = [relu]( A(MxK,bf16) @ W(KxN,bf16) + bias ) [+ resid]
//
// Block = 256 threads (8 waves), block tile 128(M) x 128(N), k-step 32.
// Wave (wr,wc) = (wave>>1, wave&1) owns a 32x64 strip = 2x4 WMMA frags.
// Double-buffered LDS pipeline: stage tile k+1 (A via
// global_load_async_to_lds_b128, W transposed via b128 read + b16 scatter)
// while the 8 WMMAs of tile k execute; then s_wait_asynccnt 0 + barrier.
// Fragment layouts used here (contiguous k for A and B, not the ISA table order):
//   A: lane(h,m): elem e -> K = h*16 + e                   (two 16B chunks)
//   B: lane(h,n): elem e -> K = h*16 + e                   (one 32B chunk)
//   C: lane(h,n): VGPR r -> row = r + 8*h
// ---------------------------------------------------------------------------
template <bool RELU, bool RESID, bool OUTF, bool OUTB>
__global__ __launch_bounds__(256) void gemm_bf16_kernel(
    const bf16_t* __restrict__ A, const bf16_t* __restrict__ W,
    const float* __restrict__ bias, const float* __restrict__ resid,
    float* __restrict__ outF, bf16_t* __restrict__ outB,
    int M, int N, int K) {
  const int lane = threadIdx.x & 31;
  const int wave = threadIdx.x >> 5;
  const int half = lane >> 4;
  const int lm   = lane & 15;
  const int wr   = wave >> 1;        // 0..3 : M strip
  const int wc   = wave & 1;         // 0..1 : N strip
  const int mblk = blockIdx.y * 128;
  const int nblk = blockIdx.x * 128;

  __shared__ __align__(16) bf16_t As[2][128 * 32];  // [m][k] row-major
  __shared__ __align__(16) bf16_t Bt[2][128 * 32];  // [n][k] transposed W tile

  // staging assignments (256 threads)
  const int tid = threadIdx.x;
  const int am = tid >> 1;            // 0..127
  const int ak = (tid & 1) * 16;      // 0 / 16
  const int bk = tid >> 3;            // 0..31
  const int bn = (tid & 7) * 16;      // 0..112

  const bf16_t* ga = A + (size_t)(mblk + am) * K + ak;   // + kt*32 per stage
  const bf16_t* gb = W + (size_t)bk * N + nblk + bn;     // + kt*32*N per stage
  const uint32_t laA = (uint32_t)(uintptr_t)(&As[0][am * 32 + ak]);
  const uint32_t ldsBufStride =
      (uint32_t)((uintptr_t)(&As[1][0]) - (uintptr_t)(&As[0][0]));

  auto stage = [&](int kt, int buf) {
    // A tile: 2 x 16B async DMA per thread
    const bf16_t* gak = ga + kt * 32;
    const uint32_t la = laA + (uint32_t)buf * ldsBufStride;
    async_copy_b128(la, gak);
    async_copy_b128(la + 16, gak + 8);
    // W tile: coalesced b128 read, transposed b16 scatter into LDS
    const bf16_t* gbk = gb + (size_t)kt * 32 * N;
    v8bf b0 = *(const v8bf*)(gbk);
    v8bf b1 = *(const v8bf*)(gbk + 8);
#pragma unroll
    for (int j = 0; j < 8; ++j) {
      Bt[buf][(bn + j) * 32 + bk]     = b0[j];
      Bt[buf][(bn + 8 + j) * 32 + bk] = b1[j];
    }
    // pull the stage-after-next toward L2
    if ((kt + 2) * 32 < K) {
      __builtin_prefetch(gak + 64, 0, 1);
      __builtin_prefetch(gbk + (size_t)64 * N, 0, 1);
    }
  };

  v8f acc[2][4] = {};

  auto compute = [&](int buf) {
    v16bf afrag[2];
#pragma unroll
    for (int mi = 0; mi < 2; ++mi) {
      const bf16_t* ar = &As[buf][(wr * 32 + mi * 16 + lm) * 32];
      afrag[mi] = cat8(*(const v8bf*)(ar + half * 16),
                       *(const v8bf*)(ar + half * 16 + 8));
    }
#pragma unroll
    for (int tn = 0; tn < 4; ++tn) {
      const v16bf bfrag =
          *(const v16bf*)(&Bt[buf][(wc * 64 + tn * 16 + lm) * 32 + half * 16]);
#pragma unroll
      for (int mi = 0; mi < 2; ++mi)
        acc[mi][tn] = __builtin_amdgcn_wmma_f32_16x16x32_bf16(
            false, afrag[mi], false, bfrag, (short)0, acc[mi][tn], false, false);
    }
  };

  const int nk = K >> 5;
  stage(0, 0);
  wait_asynccnt0();
  __syncthreads();
  for (int kt = 0; kt < nk; ++kt) {
    const int cur = kt & 1;
    if (kt + 1 < nk) stage(kt + 1, cur ^ 1);   // overlap DMA with WMMA
    compute(cur);
    wait_asynccnt0();
    __syncthreads();
  }

  // ---- epilogue ----
#pragma unroll
  for (int mi = 0; mi < 2; ++mi)
#pragma unroll
    for (int tn = 0; tn < 4; ++tn) {
      const int col = nblk + wc * 64 + tn * 16 + lm;
      const float bv = bias[col];
#pragma unroll
      for (int r = 0; r < 8; ++r) {
        const int row = mblk + wr * 32 + mi * 16 + r + 8 * half;
        float v = acc[mi][tn][r] + bv;
        if (RELU) v = fmaxf(v, 0.f);
        const size_t idx = (size_t)row * N + col;
        if (RESID) v += resid[idx];
        if (OUTF) outF[idx] = v;
        if (OUTB) outB[idx] = (bf16_t)v;
      }
    }
}

// ---------------------------------------------------------------------------
// Flash attention: one wave per 16-query tile per (b,h); kv tiles of 32.
// ---------------------------------------------------------------------------
__global__ __launch_bounds__(128) void flash_attn_kernel(
    const bf16_t* __restrict__ q, const bf16_t* __restrict__ k,
    const bf16_t* __restrict__ v, const int* __restrict__ mask,
    bf16_t* __restrict__ ctx) {
  const int lane = threadIdx.x & 31;
  const int wave = threadIdx.x >> 5;
  const int half = lane >> 4;
  const int lm   = lane & 15;
  const int bh = blockIdx.y;        // B*H = 64
  const int b  = bh / HH;
  const int h  = bh % HH;
  const int q0 = (blockIdx.x * 4 + wave) * 16;

  __shared__ __align__(16) bf16_t lds[4][16 * 32];
  bf16_t* P = lds[wave];

  const size_t base = (size_t)b * SS * DD + (size_t)h * DKK;

  // Q fragments (DK=64 -> two 16x32 A-frags)
  v16bf aQ[2];
  {
    const bf16_t* qrow = q + base + (size_t)(q0 + lm) * DD;
#pragma unroll
    for (int ks = 0; ks < 2; ++ks)
      aQ[ks] = cat8(*(const v8bf*)(qrow + ks * 32 + half * 16),
                    *(const v8bf*)(qrow + ks * 32 + half * 16 + 8));
  }

  float mrow[8], lrow[8];
#pragma unroll
  for (int r = 0; r < 8; ++r) { mrow[r] = -3.0e38f; lrow[r] = 0.f; }
  v8f acc[4] = {};

  for (int kv0 = 0; kv0 < SS; kv0 += 32) {
    // ---- scores: 16 queries x 32 keys ----
    v8f sc[2];
#pragma unroll
    for (int nt = 0; nt < 2; ++nt) {
      v8f s = {};
#pragma unroll
      for (int ks = 0; ks < 2; ++ks) {
        const bf16_t* krow =
            k + base + (size_t)(kv0 + nt * 16 + lm) * DD + ks * 32 + half * 16;
        const v16bf bK = *(const v16bf*)(krow);
        s = __builtin_amdgcn_wmma_f32_16x16x32_bf16(
            false, aQ[ks], false, bK, (short)0, s, false, false);
      }
      const int key = kv0 + nt * 16 + lm;
      const bool ok = mask[b * SS + key] != 0;
#pragma unroll
      for (int r = 0; r < 8; ++r) {
        const float val = s[r] * 0.125f;   // 1/sqrt(DK)
        s[r] = ok ? val : -1.0e9f;
      }
      sc[nt] = s;
    }

    // ---- online softmax over 16-lane halves ----
    float scale_old[8];
#pragma unroll
    for (int r = 0; r < 8; ++r) {
      float mx = fmaxf(sc[0][r], sc[1][r]);
#pragma unroll
      for (int off = 1; off < 16; off <<= 1)
        mx = fmaxf(mx, __shfl_xor(mx, off, 32));
      const float mnew = fmaxf(mrow[r], mx);
      scale_old[r] = __expf(mrow[r] - mnew);
      mrow[r] = mnew;
    }
#pragma unroll
    for (int nt = 0; nt < 2; ++nt)
#pragma unroll
      for (int r = 0; r < 8; ++r)
        sc[nt][r] = __expf(sc[nt][r] - mrow[r]);
#pragma unroll
    for (int r = 0; r < 8; ++r) {
      float sm = sc[0][r] + sc[1][r];
#pragma unroll
      for (int off = 1; off < 16; off <<= 1)
        sm += __shfl_xor(sm, off, 32);
      lrow[r] = lrow[r] * scale_old[r] + sm;
    }
#pragma unroll
    for (int t = 0; t < 4; ++t)
#pragma unroll
      for (int r = 0; r < 8; ++r) acc[t][r] *= scale_old[r];

    // ---- P: C-layout -> A-layout via LDS ----
#pragma unroll
    for (int nt = 0; nt < 2; ++nt)
#pragma unroll
      for (int r = 0; r < 8; ++r)
        P[(r + 8 * half) * 32 + nt * 16 + lm] = (bf16_t)sc[nt][r];
    __syncthreads();
    v16bf pa;
    {
      const bf16_t* prow = P + lm * 32;
      pa = cat8(*(const v8bf*)(prow + half * 16),
                *(const v8bf*)(prow + half * 16 + 8));
    }

    // ---- acc += P(16x32) @ V(32x64) ----
#pragma unroll
    for (int dt = 0; dt < 4; ++dt) {
      v16bf bV;
      const bf16_t* vcol =
          v + base + (size_t)(kv0 + half * 16) * DD + dt * 16 + lm;
#pragma unroll
      for (int e = 0; e < 16; ++e) bV[e] = vcol[(size_t)e * DD];
      acc[dt] = __builtin_amdgcn_wmma_f32_16x16x32_bf16(
          false, pa, false, bV, (short)0, acc[dt], false, false);
    }
    __syncthreads();
  }

  // ---- normalize and store ctx bf16 at (B*S, D) ----
#pragma unroll
  for (int dt = 0; dt < 4; ++dt)
#pragma unroll
    for (int r = 0; r < 8; ++r) {
      const int row = q0 + r + 8 * half;
      const float o = acc[dt][r] / lrow[r];
      ctx[base + (size_t)row * DD + dt * 16 + lm] = (bf16_t)o;
    }
}

// ---------------------------------------------------------------------------
// Launch
// ---------------------------------------------------------------------------
extern "C" void kernel_launch(void* const* d_in, const int* in_sizes, int n_in,
                              void* d_out, int out_size, void* d_ws,
                              size_t ws_size, hipStream_t stream) {
  (void)in_sizes; (void)n_in; (void)out_size; (void)ws_size;
  const float* x        = (const float*)d_in[0];
  const int*   src_mask = (const int*)  d_in[1];
  const float* wq = (const float*)d_in[2];
  const float* bq = (const float*)d_in[3];
  const float* wk = (const float*)d_in[4];
  const float* bk = (const float*)d_in[5];
  const float* wv = (const float*)d_in[6];
  const float* bv = (const float*)d_in[7];
  const float* wo = (const float*)d_in[8];
  const float* bo = (const float*)d_in[9];
  const float* w1 = (const float*)d_in[10];
  const float* b1 = (const float*)d_in[11];
  const float* w2 = (const float*)d_in[12];
  const float* b2 = (const float*)d_in[13];
  const float* alpha1 = (const float*)d_in[14];
  const float* beta1  = (const float*)d_in[15];
  const float* alpha2 = (const float*)d_in[16];
  const float* beta2  = (const float*)d_in[17];
  float* out = (float*)d_out;

  // workspace carve-up
  char* ws = (char*)d_ws;
  size_t off = 0;
  auto take = [&](size_t bytes) -> char* {
    char* p = ws + off;
    off += (bytes + 255) & ~(size_t)255;
    return p;
  };
  bf16_t* wq_b  = (bf16_t*)take((size_t)DD * DD * 2);
  bf16_t* wk_b  = (bf16_t*)take((size_t)DD * DD * 2);
  bf16_t* wv_b  = (bf16_t*)take((size_t)DD * DD * 2);
  bf16_t* wo_b  = (bf16_t*)take((size_t)DD * DD * 2);
  bf16_t* w1_b  = (bf16_t*)take((size_t)DD * DFFF * 2);
  bf16_t* w2_b  = (bf16_t*)take((size_t)DFFF * DD * 2);
  bf16_t* xn_b  = (bf16_t*)take((size_t)MROWS * DD * 2);
  bf16_t* q_b   = (bf16_t*)take((size_t)MROWS * DD * 2);
  bf16_t* k_b   = (bf16_t*)take((size_t)MROWS * DD * 2);
  bf16_t* v_b   = (bf16_t*)take((size_t)MROWS * DD * 2);
  bf16_t* ctx_b = (bf16_t*)take((size_t)MROWS * DD * 2);
  float*  x1    = (float*) take((size_t)MROWS * DD * 4);
  bf16_t* xn2_b = (bf16_t*)take((size_t)MROWS * DD * 2);
  bf16_t* h1_b  = (bf16_t*)take((size_t)MROWS * DFFF * 2);

  // 1) weights -> bf16
  auto cvt = [&](const float* src, bf16_t* dst, int n) {
    cvt_f32_bf16_kernel<<<(n + 255) / 256, 256, 0, stream>>>(src, dst, n);
  };
  cvt(wq, wq_b, DD * DD);
  cvt(wk, wk_b, DD * DD);
  cvt(wv, wv_b, DD * DD);
  cvt(wo, wo_b, DD * DD);
  cvt(w1, w1_b, DD * DFFF);
  cvt(w2, w2_b, DFFF * DD);

  // 2) LN1
  layernorm_kernel<<<MROWS, 256, 0, stream>>>(x, alpha1, beta1, xn_b);

  // 3) QKV projections (bf16 out)
  dim3 blk(256);
  dim3 gQ(DD / 128, MROWS / 128);
  gemm_bf16_kernel<false, false, false, true><<<gQ, blk, 0, stream>>>(
      xn_b, wq_b, bq, nullptr, nullptr, q_b, MROWS, DD, DD);
  gemm_bf16_kernel<false, false, false, true><<<gQ, blk, 0, stream>>>(
      xn_b, wk_b, bk, nullptr, nullptr, k_b, MROWS, DD, DD);
  gemm_bf16_kernel<false, false, false, true><<<gQ, blk, 0, stream>>>(
      xn_b, wv_b, bv, nullptr, nullptr, v_b, MROWS, DD, DD);

  // 4) flash attention -> ctx (B*S, D) bf16
  flash_attn_kernel<<<dim3(SS / 64, BB * HH), dim3(128), 0, stream>>>(
      q_b, k_b, v_b, src_mask, ctx_b);

  // 5) x1 = x + ctx @ wo + bo   (f32)
  gemm_bf16_kernel<false, true, true, false><<<gQ, blk, 0, stream>>>(
      ctx_b, wo_b, bo, x, x1, nullptr, MROWS, DD, DD);

  // 6) LN2
  layernorm_kernel<<<MROWS, 256, 0, stream>>>(x1, alpha2, beta2, xn2_b);

  // 7) h1 = relu(xn2 @ w1 + b1)  (bf16)
  dim3 gF(DFFF / 128, MROWS / 128);
  gemm_bf16_kernel<true, false, false, true><<<gF, blk, 0, stream>>>(
      xn2_b, w1_b, b1, nullptr, nullptr, h1_b, MROWS, DFFF, DD);

  // 8) out = x1 + h1 @ w2 + b2  (f32 final)
  gemm_bf16_kernel<false, true, true, false><<<gQ, blk, 0, stream>>>(
      h1_b, w2_b, b2, x1, out, nullptr, MROWS, DD, DFFF);
}
